// GENConv_Classifier_63419487092761
// MI455X (gfx1250) — matrix operand order repaired and checked by reading, not compiled, not run on hardware
//
#include <hip/hip_runtime.h>
#include <cstddef>
#include <cstdint>

typedef __attribute__((ext_vector_type(16))) _Float16 v16h;
typedef __attribute__((ext_vector_type(8)))  float    v8f;

#define NNODES  100000
#define NEDGES  600000
#define NGRAPHS 512
#define DIN     48
#define DINP    64      // DIN padded to K-step multiple (32)
#define W128    128
#define BN_EPSF 1e-5f
#define GEN_EPSF 1e-7f

// ---------------- BatchNorm: per-feature sum / sumsq ----------------
__global__ void bn_stats_kernel(const float* __restrict__ X, float* __restrict__ acc,
                                int N, int F, int rowsPerBlock) {
  __shared__ float s_sum[256];
  __shared__ float s_sq[256];
  int t = threadIdx.x;
  s_sum[t] = 0.f;
  s_sq[t]  = 0.f;
  __syncthreads();
  long r0 = (long)blockIdx.x * rowsPerBlock;
  int elems = rowsPerBlock * F;
  for (int i = t; i < elems; i += blockDim.x) {
    long r = r0 + i / F;
    int  f = i % F;
    if (r < N) {
      float v = X[r * (long)F + f];
      atomicAdd(&s_sum[f], v);
      atomicAdd(&s_sq[f],  v * v);
    }
  }
  __syncthreads();
  if (t < F) {
    atomicAdd(&acc[t],     s_sum[t]);
    atomicAdd(&acc[F + t], s_sq[t]);
  }
}

// Normalize; input stride F, output stride Fout (pad columns written as 0).
__global__ void bn_apply_kernel(const float* __restrict__ Xin, float* __restrict__ Xout,
                                const float* __restrict__ acc,
                                const float* __restrict__ g, const float* __restrict__ b,
                                int N, int F, int Fout, int relu) {
  long idx = (long)blockIdx.x * blockDim.x + threadIdx.x;
  long total = (long)N * Fout;
  if (idx >= total) return;
  long r = idx / Fout;
  int  f = (int)(idx % Fout);
  float y = 0.f;
  if (f < F) {
    float invN = 1.0f / (float)N;
    float mean = acc[f] * invN;
    float var  = acc[F + f] * invN - mean * mean;
    y = g[f] * (Xin[r * (long)F + f] - mean) * rsqrtf(var + BN_EPSF) + b[f];
    if (relu) y = fmaxf(y, 0.f);
  }
  Xout[idx] = y;
}

// ---------------- scatter-mean of bn(X) over graph id ----------------
__global__ void scatter_sum_kernel(const float* __restrict__ Xn, const int* __restrict__ batch,
                                   float* __restrict__ gsum, float* __restrict__ gcnt) {
  long idx = (long)blockIdx.x * blockDim.x + threadIdx.x;
  long total = (long)NNODES * DIN;
  if (idx >= total) return;
  int i = (int)(idx / DIN);
  int f = (int)(idx % DIN);
  int g = batch[i];
  atomicAdd(&gsum[g * DIN + f], Xn[(size_t)i * DINP + f]);
  if (f == 0) atomicAdd(&gcnt[g], 1.0f);
}

// writes x3 padded to stride DINP (pad cols = 0)
__global__ void scatter_div_kernel(float* __restrict__ x3, const float* __restrict__ gsum,
                                   const float* __restrict__ gcnt) {
  int idx = blockIdx.x * blockDim.x + threadIdx.x;
  if (idx >= NGRAPHS * DINP) return;
  int g = idx / DINP;
  int f = idx % DINP;
  x3[idx] = (f < DIN) ? gsum[g * DIN + f] / fmaxf(gcnt[g], 1.0f) : 0.f;
}

// ---------------- weight packing into B-fragment layout ----------------
// Wp layout: fragment (kc, ct) holds 32 lanes x 16 halfs; lane = half*16 + (col-ct*16),
// element e -> K = kc*32 + 8*half + (e&7) + 16*(e>>3).  Zero-pads K in [Kin, Kpad).
__global__ void pack_w_kernel(const float* __restrict__ W, _Float16* __restrict__ Wp,
                              int Kin, int Kpad, int K2) {
  long idx = (long)blockIdx.x * blockDim.x + threadIdx.x;
  long total = (long)Kpad * K2;
  if (idx >= total) return;
  int e    = (int)(idx & 15);
  int lane = (int)((idx >> 4) & 31);
  long rest = idx >> 9;
  int nct = K2 >> 4;
  int ct = (int)(rest % nct);
  int kc = (int)(rest / nct);
  int half = lane >> 4;
  int k   = kc * 32 + 8 * half + (e & 7) + 16 * (e >> 3);
  int col = ct * 16 + (lane & 15);
  Wp[idx] = (k < Kin) ? (_Float16)W[(size_t)k * K2 + col] : (_Float16)0.f;
}

// ---------------- WMMA GEMM: C[N,K2] = act(actIn(A[N,K1]) @ W + bias) ----------------
// One wave per block; each wave computes a 16x64 output tile (4 column tiles, 4 accs).
// A fragment loaded once per K-chunk and reused by 4 independent back-to-back WMMAs.
// N % 16 == 0, K1 % 32 == 0, K2 % 64 == 0 guaranteed => no guards, EXEC all-ones.
template <bool RELU_IN, bool RELU_OUT>
__global__ __launch_bounds__(32)
void gemm_wmma_kernel(const float* __restrict__ A, const _Float16* __restrict__ Wp,
                      const float* __restrict__ bias, float* __restrict__ C,
                      int K1, int K2) {
  int lane = threadIdx.x & 31;
  int row0 = blockIdx.x * 16;
  int ct0  = blockIdx.y * 4;      // 4 column tiles per wave
  int half = lane >> 4;
  int lidx = lane & 15;
  const float* Arow = A + (size_t)(row0 + lidx) * K1 + 8 * half;
  const v16h* Wv = (const v16h*)Wp + (size_t)ct0 * 32 + lane;
  int nct = K2 >> 4;
  int nkc = K1 >> 5;
  v8f acc0 = {}, acc1 = {}, acc2 = {}, acc3 = {};
  for (int kc = 0; kc < nkc; ++kc) {
    int kk = kc << 5;
    float4 a0 = *(const float4*)(Arow + kk);
    float4 a1 = *(const float4*)(Arow + kk + 4);
    float4 a2 = *(const float4*)(Arow + kk + 16);
    float4 a3 = *(const float4*)(Arow + kk + 20);
    if (RELU_IN) {
      a0.x = fmaxf(a0.x, 0.f); a0.y = fmaxf(a0.y, 0.f); a0.z = fmaxf(a0.z, 0.f); a0.w = fmaxf(a0.w, 0.f);
      a1.x = fmaxf(a1.x, 0.f); a1.y = fmaxf(a1.y, 0.f); a1.z = fmaxf(a1.z, 0.f); a1.w = fmaxf(a1.w, 0.f);
      a2.x = fmaxf(a2.x, 0.f); a2.y = fmaxf(a2.y, 0.f); a2.z = fmaxf(a2.z, 0.f); a2.w = fmaxf(a2.w, 0.f);
      a3.x = fmaxf(a3.x, 0.f); a3.y = fmaxf(a3.y, 0.f); a3.z = fmaxf(a3.z, 0.f); a3.w = fmaxf(a3.w, 0.f);
    }
    v16h a;
    a[0]  = (_Float16)a0.x; a[1]  = (_Float16)a0.y; a[2]  = (_Float16)a0.z; a[3]  = (_Float16)a0.w;
    a[4]  = (_Float16)a1.x; a[5]  = (_Float16)a1.y; a[6]  = (_Float16)a1.z; a[7]  = (_Float16)a1.w;
    a[8]  = (_Float16)a2.x; a[9]  = (_Float16)a2.y; a[10] = (_Float16)a2.z; a[11] = (_Float16)a2.w;
    a[12] = (_Float16)a3.x; a[13] = (_Float16)a3.y; a[14] = (_Float16)a3.z; a[15] = (_Float16)a3.w;
    const v16h* Wk = Wv + (size_t)kc * nct * 32;
    v16h b0 = Wk[0];
    v16h b1 = Wk[32];
    v16h b2 = Wk[64];
    v16h b3 = Wk[96];
    acc0 = __builtin_amdgcn_wmma_f32_16x16x32_f16(false, a, false, b0, (short)0, acc0, false, false);
    acc1 = __builtin_amdgcn_wmma_f32_16x16x32_f16(false, a, false, b1, (short)0, acc1, false, false);
    acc2 = __builtin_amdgcn_wmma_f32_16x16x32_f16(false, a, false, b2, (short)0, acc2, false, false);
    acc3 = __builtin_amdgcn_wmma_f32_16x16x32_f16(false, a, false, b3, (short)0, acc3, false, false);
  }
  float* Crow = C + (size_t)(row0 + 8 * half) * K2 + ct0 * 16 + lidx;
#pragma unroll
  for (int t = 0; t < 4; ++t) {
    v8f accv = (t == 0) ? acc0 : (t == 1) ? acc1 : (t == 2) ? acc2 : acc3;
    float bv = bias[(ct0 + t) * 16 + lidx];
    float* Cp = Crow + t * 16;
#pragma unroll
    for (int v = 0; v < 8; ++v) {
      float r = accv[v] + bv;
      if (RELU_OUT) r = fmaxf(r, 0.f);
      Cp[(size_t)v * K2] = r;
    }
  }
}

// ---------------- GENConv softmax aggregation (3 edge passes) ----------------
// logits > 0 always (msg >= GEN_EPS, t = 1), so uint-bit atomicMax == float max,
// and zero-init mx reproduces the reference's "empty segment -> 0" rule.
__global__ void edge_max_kernel(const float* __restrict__ x1, const int* __restrict__ src,
                                const int* __restrict__ dst, const float* __restrict__ tptr,
                                unsigned int* __restrict__ mx) {
  int e = blockIdx.x;
  int f = threadIdx.x;
  int s = src[e], d = dst[e];
  float m = fmaxf(x1[(size_t)s * W128 + f], 0.f) + GEN_EPSF;
  float logit = m * tptr[0];
  atomicMax(&mx[(size_t)d * W128 + f], __float_as_uint(logit));
}

__global__ void edge_denom_kernel(const float* __restrict__ x1, const int* __restrict__ src,
                                  const int* __restrict__ dst, const float* __restrict__ tptr,
                                  const float* __restrict__ mx, float* __restrict__ denom) {
  int e = blockIdx.x;
  int f = threadIdx.x;
  int s = src[e], d = dst[e];
  float m = fmaxf(x1[(size_t)s * W128 + f], 0.f) + GEN_EPSF;
  float logit = m * tptr[0];
  float ex = __expf(logit - mx[(size_t)d * W128 + f]);
  atomicAdd(&denom[(size_t)d * W128 + f], ex);
}

__global__ void edge_agg_kernel(const float* __restrict__ x1, const int* __restrict__ src,
                                const int* __restrict__ dst, const float* __restrict__ tptr,
                                const float* __restrict__ mx, const float* __restrict__ denom,
                                float* __restrict__ hg) {
  int e = blockIdx.x;
  int f = threadIdx.x;
  int s = src[e], d = dst[e];
  float m = fmaxf(x1[(size_t)s * W128 + f], 0.f) + GEN_EPSF;
  float logit = m * tptr[0];
  size_t o = (size_t)d * W128 + f;
  float ex = __expf(logit - mx[o]);
  atomicAdd(&hg[o], m * ex / (denom[o] + 1e-16f));
}

// ---------------- output head: out[g] = x4[g,:] . oW + ob ----------------
__global__ void head_kernel(const float* __restrict__ x4, const float* __restrict__ oW,
                            const float* __restrict__ ob, float* __restrict__ out) {
  __shared__ float red[W128];
  int g = blockIdx.x;
  int t = threadIdx.x;
  red[t] = x4[(size_t)g * W128 + t] * oW[t];
  __syncthreads();
  for (int s = 64; s > 0; s >>= 1) {
    if (t < s) red[t] += red[t + s];
    __syncthreads();
  }
  if (t == 0) out[g] = red[0] + ob[0];
}

extern "C" void kernel_launch(void* const* d_in, const int* in_sizes, int n_in,
                              void* d_out, int out_size, void* d_ws, size_t ws_size,
                              hipStream_t stream) {
  const float* X     = (const float*)d_in[0];
  const int*   ei    = (const int*)  d_in[1];
  const int*   src   = ei;
  const int*   dst   = ei + NEDGES;
  const int*   batch = (const int*)  d_in[2];
  const float* bn_g  = (const float*)d_in[3];
  const float* bn_b  = (const float*)d_in[4];
  const float* n1_W1 = (const float*)d_in[5];
  const float* n1_b1 = (const float*)d_in[6];
  const float* n1_W2 = (const float*)d_in[7];
  const float* n1_b2 = (const float*)d_in[8];
  const float* n1_W3 = (const float*)d_in[9];
  const float* n1_b3 = (const float*)d_in[10];
  const float* c_W1  = (const float*)d_in[11];
  const float* c_b1  = (const float*)d_in[12];
  const float* c_bn_g = (const float*)d_in[13];
  const float* c_bn_b = (const float*)d_in[14];
  const float* c_W2  = (const float*)d_in[15];
  const float* c_b2  = (const float*)d_in[16];
  const float* tpar  = (const float*)d_in[17];
  const float* n2_W1 = (const float*)d_in[18];
  const float* n2_b1 = (const float*)d_in[19];
  const float* n2_W2 = (const float*)d_in[20];
  const float* n2_b2 = (const float*)d_in[21];
  const float* n2_W3 = (const float*)d_in[22];
  const float* n2_b3 = (const float*)d_in[23];
  const float* n2_W4 = (const float*)d_in[24];
  const float* n2_b4 = (const float*)d_in[25];
  const float* o_W   = (const float*)d_in[26];
  const float* o_b   = (const float*)d_in[27];

  float* out = (float*)d_out;          // [512]
  float* x2  = out + NGRAPHS;          // [100000,128]

  // ---- workspace carve-up ----
  float* p = (float*)d_ws;
  float* Xn  = p;  p += (size_t)NNODES * DINP;      // bn(X), padded stride 64
  float* S1  = p;  p += (size_t)NNODES * W128;      // H1, then mx
  float* S2  = p;  p += (size_t)NNODES * W128;      // H2, then denom
  float* x1  = p;  p += (size_t)NNODES * W128;
  float* hg  = p;  p += (size_t)NNODES * W128;      // agg + x1
  float* T   = p;  p += (size_t)NNODES * 2 * W128;  // conv MLP hidden, BN in place
  float* acc48  = p; p += 2 * DIN;
  float* acc256 = p; p += 4 * W128;
  float* gsum   = p; p += NGRAPHS * DIN;
  float* gcnt   = p; p += NGRAPHS;
  float* x3     = p; p += NGRAPHS * DINP;           // padded stride 64
  float* hs1    = p; p += NGRAPHS * W128;
  float* hs2    = p; p += NGRAPHS * W128;
  // packed f16 weights (32B-aligned for v16h loads)
  _Float16* wp = (_Float16*)(((uintptr_t)p + 31) & ~(uintptr_t)31);
  _Float16* Wp_n1_1 = wp; wp += DINP * W128;
  _Float16* Wp_n1_2 = wp; wp += W128 * W128;
  _Float16* Wp_n1_3 = wp; wp += W128 * W128;
  _Float16* Wp_c1   = wp; wp += W128 * 2 * W128;
  _Float16* Wp_c2   = wp; wp += 2 * W128 * W128;
  _Float16* Wp_n2_1 = wp; wp += DINP * W128;
  _Float16* Wp_n2_2 = wp; wp += W128 * W128;
  _Float16* Wp_n2_3 = wp; wp += W128 * W128;
  _Float16* Wp_n2_4 = wp; wp += W128 * W128;

  dim3 blk(32);                                   // one wave per block
  dim3 gN128(NNODES / 16, W128 / 64);             // (6250, 2)
  dim3 gN256(NNODES / 16, (2 * W128) / 64);       // (6250, 4)
  dim3 gG128(NGRAPHS / 16, W128 / 64);            // (32, 2)

  // 0) pack all weights into f16 B-fragment layout
  {
    auto pk = [&](const float* W, _Float16* Wpk, int Kin, int Kpad, int K2) {
      long tot = (long)Kpad * K2;
      pack_w_kernel<<<(int)((tot + 255) / 256), 256, 0, stream>>>(W, Wpk, Kin, Kpad, K2);
    };
    pk(n1_W1, Wp_n1_1, DIN,      DINP,     W128);
    pk(n1_W2, Wp_n1_2, W128,     W128,     W128);
    pk(n1_W3, Wp_n1_3, W128,     W128,     W128);
    pk(c_W1,  Wp_c1,   W128,     W128, 2 * W128);
    pk(c_W2,  Wp_c2,   2 * W128, 2 * W128, W128);
    pk(n2_W1, Wp_n2_1, DIN,      DINP,     W128);
    pk(n2_W2, Wp_n2_2, W128,     W128,     W128);
    pk(n2_W3, Wp_n2_3, W128,     W128,     W128);
    pk(n2_W4, Wp_n2_4, W128,     W128,     W128);
  }

  // 1) input BN -> Xn (padded stride 64)
  hipMemsetAsync(acc48, 0, sizeof(float) * 2 * DIN, stream);
  bn_stats_kernel<<<(NNODES + 255) / 256, 256, 0, stream>>>(X, acc48, NNODES, DIN, 256);
  {
    long total = (long)NNODES * DINP;
    bn_apply_kernel<<<(int)((total + 255) / 256), 256, 0, stream>>>(X, Xn, acc48, bn_g, bn_b,
                                                                    NNODES, DIN, DINP, 0);
  }

  // 2) scatter-mean of Xn -> x3 (padded stride 64)
  hipMemsetAsync(gsum, 0, sizeof(float) * NGRAPHS * DIN, stream);
  hipMemsetAsync(gcnt, 0, sizeof(float) * NGRAPHS, stream);
  {
    long total = (long)NNODES * DIN;
    scatter_sum_kernel<<<(int)((total + 255) / 256), 256, 0, stream>>>(Xn, batch, gsum, gcnt);
    scatter_div_kernel<<<(NGRAPHS * DINP + 255) / 256, 256, 0, stream>>>(x3, gsum, gcnt);
  }

  // 3) nn1: relu(Xn) -> W1 -> relu -> W2 -> relu -> W3
  gemm_wmma_kernel<true,  true ><<<gN128, blk, 0, stream>>>(Xn, Wp_n1_1, n1_b1, S1, DINP, W128);
  gemm_wmma_kernel<false, true ><<<gN128, blk, 0, stream>>>(S1, Wp_n1_2, n1_b2, S2, W128, W128);
  gemm_wmma_kernel<false, false><<<gN128, blk, 0, stream>>>(S2, Wp_n1_3, n1_b3, x1, W128, W128);

  // 4) GENConv softmax aggregation; hg starts as x1 so final hg = agg + x1
  hipMemsetAsync(S1, 0, sizeof(float) * (size_t)NNODES * W128, stream);  // mx
  hipMemsetAsync(S2, 0, sizeof(float) * (size_t)NNODES * W128, stream);  // denom
  hipMemcpyAsync(hg, x1, sizeof(float) * (size_t)NNODES * W128, hipMemcpyDeviceToDevice, stream);
  edge_max_kernel  <<<NEDGES, W128, 0, stream>>>(x1, src, dst, tpar, (unsigned int*)S1);
  edge_denom_kernel<<<NEDGES, W128, 0, stream>>>(x1, src, dst, tpar, S1, S2);
  edge_agg_kernel  <<<NEDGES, W128, 0, stream>>>(x1, src, dst, tpar, S1, S2, hg);

  // 5) conv MLP: hg @ c_W1 -> BN(256)+relu (in place) -> c_W2 -> x2 (into d_out)
  gemm_wmma_kernel<false, false><<<gN256, blk, 0, stream>>>(hg, Wp_c1, c_b1, T, W128, 2 * W128);
  hipMemsetAsync(acc256, 0, sizeof(float) * 4 * W128, stream);
  bn_stats_kernel<<<(NNODES + 127) / 128, 256, 0, stream>>>(T, acc256, NNODES, 2 * W128, 128);
  {
    long total = (long)NNODES * 2 * W128;
    bn_apply_kernel<<<(int)((total + 255) / 256), 256, 0, stream>>>(T, T, acc256, c_bn_g, c_bn_b,
                                                                    NNODES, 2 * W128, 2 * W128, 1);
  }
  gemm_wmma_kernel<false, false><<<gN128, blk, 0, stream>>>(T, Wp_c2, c_b2, x2, 2 * W128, W128);

  // 6) nn2 on x3 [512,64(pad)] + output head
  gemm_wmma_kernel<false, true ><<<gG128, blk, 0, stream>>>(x3,  Wp_n2_1, n2_b1, hs1, DINP, W128);
  gemm_wmma_kernel<false, true ><<<gG128, blk, 0, stream>>>(hs1, Wp_n2_2, n2_b2, hs2, W128, W128);
  gemm_wmma_kernel<false, true ><<<gG128, blk, 0, stream>>>(hs2, Wp_n2_3, n2_b3, hs1, W128, W128);
  gemm_wmma_kernel<false, false><<<gG128, blk, 0, stream>>>(hs1, Wp_n2_4, n2_b4, hs2, W128, W128);
  head_kernel<<<NGRAPHS, W128, 0, stream>>>(hs2, o_W, o_b, out);
}
